// NyanEncoder_257698038442
// MI455X (gfx1250) — matrix operand order repaired and checked by reading, not compiled, not run on hardware
//
#include <hip/hip_runtime.h>
#include <hip/hip_bf16.h>

// ---------------- problem constants (from reference) ----------------
#define FDIM 32      // node input features
#define SDIM 6       // edge features
#define HDIM 32      // hidden width
#define LATD 64      // latent
#define PREW 16      // precondition width
#define ALPHA 0.05f

typedef __attribute__((ext_vector_type(8)))  float  v8f;
typedef __attribute__((ext_vector_type(16))) __bf16 v16bf;

union Frag16 {
    v16bf v;
    unsigned short u[16];
    uint4 q[2];
};

__device__ __forceinline__ float leaky(float x) { return x > 0.f ? x : ALPHA * x; }

__device__ __forceinline__ unsigned short f32_to_bf16(float x) {
    unsigned int u = __float_as_uint(x);
    unsigned int r = 0x7fffu + ((u >> 16) & 1u);   // round to nearest even
    return (unsigned short)((u + r) >> 16);
}

// load a 32-byte fragment as two 16-byte vector loads (ds_load_b128 / global_load_b128)
__device__ __forceinline__ void ld_frag(Frag16& f, const unsigned short* p) {
    const uint4* q = (const uint4*)p;
    f.q[0] = q[0];
    f.q[1] = q[1];
}

// ---------------- generic zero ----------------
__global__ void zero_f32(float* p, int n) {
    int i = blockIdx.x * blockDim.x + threadIdx.x;
    if (i < n) p[i] = 0.f;
}

// ---------------- build W' pre-permuted into B-fragment order ----------------
// Logical W' = [K, HDIM] with K = S*fin rows of Wk, fin rows of bk, zero pad.
// Output layout: Wfrag[tile][chunk][lane][16] bf16 (tile selects N columns 0-15 / 16-31).
// B 32x16 layout assumed: N = lane&15, half = lane>>4, VGPR vv holds K = 2*(vv&3) + 16*(vv>>2) + 8*half (+bit).
__global__ void build_wb(const float* __restrict__ Wk, const float* __restrict__ bk,
                         unsigned short* __restrict__ Wfrag, int fin, int nChunks) {
    int i = blockIdx.x * blockDim.x + threadIdx.x;  // ushort slot
    int total = 2 * nChunks * 32 * 16;
    if (i >= total) return;
    int idx   = i & 15;
    int lane  = (i >> 4) & 31;
    int rest  = i >> 9;
    int chunk = rest % nChunks;
    int tile  = rest / nChunks;
    int n     = (lane & 15) + 16 * tile;
    int half  = lane >> 4;
    int vv    = idx >> 1, bit = idx & 1;
    int k     = chunk * 32 + 2 * (vv & 3) + 16 * (vv >> 2) + 8 * half + bit;
    int sfin  = SDIM * fin;
    float v = 0.f;
    if (k < sfin)             v = Wk[(size_t)(k / fin) * (fin * HDIM) + (k % fin) * HDIM + n];
    else if (k < sfin + fin)  v = bk[(k - sfin) * HDIM + n];
    Wfrag[i] = f32_to_bf16(v);
}

// ---------------- precondition: h0 = leaky(x @ W_pre + b_pre), [N,16] ----------------
__global__ __launch_bounds__(256)
void pre_kernel(const float* __restrict__ x, const float* __restrict__ W,
                const float* __restrict__ b, float* __restrict__ h0, int Ncnt) {
    __shared__ float w_s[FDIM * PREW];
    __shared__ float b_s[PREW];
    int t = threadIdx.x;
    for (int i = t; i < FDIM * PREW; i += blockDim.x) w_s[i] = W[i];
    if (t < PREW) b_s[t] = b[t];
    __syncthreads();
    int n = blockIdx.x * blockDim.x + t;
    if (n >= Ncnt) return;
    float acc[PREW];
    #pragma unroll
    for (int c = 0; c < PREW; ++c) acc[c] = b_s[c];
    #pragma unroll
    for (int f = 0; f < FDIM; ++f) {
        float xv = x[(size_t)n * FDIM + f];
        #pragma unroll
        for (int c = 0; c < PREW; ++c) acc[c] += xv * w_s[f * PREW + c];
    }
    #pragma unroll
    for (int c = 0; c < PREW; ++c) h0[(size_t)n * PREW + c] = leaky(acc[c]);
}

// ---------------- ECC edge messages via WMMA + scatter atomics ----------------
// msg[16e x 32c] = A(Z)[16 x K] x B(W')[K x 32] with f32 accumulation.
// Z is built straight into A-fragment order in LDS -> phase 3 loads 2x ds_load_b128 per A
// fragment and 2x global_load_b128 per (L2-resident) B fragment.
// 4 waves / block, 16 edges / wave, 64 edges / block.
__global__ __launch_bounds__(128)
void ecc_edge_wmma(const float* __restrict__ xin, int finLog2, int nChunks,
                   const float* __restrict__ eattr,
                   const int* __restrict__ src, const int* __restrict__ tgt,
                   const unsigned short* __restrict__ Wfrag,  // [2][nChunks][32][16] bf16
                   float* __restrict__ agg, int E_total) {
    constexpr int EPB = 64;   // edges per block
    const int fin  = 1 << finLog2;
    const int Kpad = nChunks << 5;
    extern __shared__ unsigned char smem[];
    unsigned short* zfrag = (unsigned short*)smem;                      // [4][nChunks][32][16]
    float* xs             = (float*)(smem + (size_t)EPB * Kpad * 2);    // [EPB][fin]
    float* es             = xs + EPB * fin;                             // [EPB][SDIM]
    int*   tg             = (int*)(es + EPB * SDIM);                    // [EPB]

    const int tid  = threadIdx.x;
    const int base = blockIdx.x * EPB;

    if (tid == 0) __builtin_prefetch(Wfrag, 0, 0);   // global_prefetch_b8 on W'

    // ---- phase 1: stage tgt, e-rows, gathered x-rows into LDS ----
    for (int i = tid; i < EPB; i += blockDim.x) {
        int eidx = base + i;
        tg[i] = (eidx < E_total) ? tgt[eidx] : -1;
    }
    for (int i = tid; i < EPB * SDIM; i += blockDim.x) {
        int el = i / SDIM, s = i % SDIM, eidx = base + el;
        es[i] = (eidx < E_total) ? eattr[(size_t)eidx * SDIM + s] : 0.f;
    }
    for (int i = tid; i < (EPB << finLog2); i += blockDim.x) {
        int el = i >> finLog2, f = i & (fin - 1), eidx = base + el;
        xs[i] = (eidx < E_total) ? xin[((size_t)src[eidx] << finLog2) + f] : 0.f;
    }
    __syncthreads();

    // ---- phase 2: build Z directly in A-fragment order (packed b32 LDS stores) ----
    // ushort slot i = ((g*nChunks + chunk)*32 + lane)*16 + 2*vv + bit
    // inverse of A 16x32 layout: k32 = 2*(vv&3) + 16*(vv>>2) + 8*half + bit, M = lane&15
    const int sfin = SDIM << finLog2;
    const int totalPairs = (EPB * Kpad) >> 1;
    for (int p = tid; p < totalPairs; p += blockDim.x) {
        int vv    = p & 7;
        int lane  = (p >> 3) & 31;
        int rest  = p >> 8;
        int chunk = rest % nChunks;
        int g     = rest / nChunks;
        int m     = lane & 15, half = lane >> 4;
        int k     = chunk * 32 + 2 * (vv & 3) + 16 * (vv >> 2) + 8 * half;
        int el    = g * 16 + m;
        unsigned int pack = 0;
        #pragma unroll
        for (int bit = 0; bit < 2; ++bit) {
            int kk = k + bit;
            float v = 0.f;
            if (kk < sfin)            v = es[el * SDIM + (kk >> finLog2)] * xs[(el << finLog2) + (kk & (fin - 1))];
            else if (kk < sfin + fin) v = xs[(el << finLog2) + (kk - sfin)];
            pack |= (unsigned int)f32_to_bf16(v) << (16 * bit);
        }
        ((unsigned int*)zfrag)[p] = pack;
    }
    __syncthreads();

    // ---- phase 3: per-wave WMMA over K chunks, two N=16 tiles ----
    const int wave = tid >> 5;
    const int lane = tid & 31;
    const int m    = lane & 15;
    const int half = lane >> 4;
    const unsigned short* zw = zfrag + (((size_t)wave * nChunks * 32 + lane) << 4);
    const unsigned short* w0 = Wfrag + ((size_t)lane << 4);
    const unsigned short* w1 = w0 + ((size_t)nChunks << 9);

    v8f acc0 = {}; v8f acc1 = {};
    for (int chunk = 0; chunk < nChunks; ++chunk) {
        Frag16 a, b0, b1;
        ld_frag(a,  zw + ((size_t)chunk << 9));   // 2x ds_load_b128
        ld_frag(b0, w0 + ((size_t)chunk << 9));   // 2x global_load_b128 (L2 hit)
        ld_frag(b1, w1 + ((size_t)chunk << 9));
        acc0 = __builtin_amdgcn_wmma_f32_16x16x32_bf16(false, a.v, false, b0.v,
                                                       (short)0, acc0, false, false);
        acc1 = __builtin_amdgcn_wmma_f32_16x16x32_bf16(false, a.v, false, b1.v,
                                                       (short)0, acc1, false, false);
    }

    // ---- scatter-add into agg (segment_sum over tgt) ----
    // D layout: VGPR j -> edge row (j + 8*half), col = m (+16 for tile 1)
    #pragma unroll
    for (int j = 0; j < 8; ++j) {
        int row = j + 8 * half;
        int t   = tg[wave * 16 + row];
        if (t >= 0) {
            atomicAdd(&agg[(size_t)t * HDIM + m],      acc0[j]);
            atomicAdd(&agg[(size_t)t * HDIM + 16 + m], acc1[j]);
        }
    }
}

// ---------------- node update: h' = leaky(agg + h @ Wr + b), one thread per node ----------
__global__ __launch_bounds__(256)
void node_update(const float* __restrict__ hin, const float* __restrict__ agg,
                 const float* __restrict__ Wr, const float* __restrict__ b,
                 float* __restrict__ hout, int fin, int Ncnt) {
    __shared__ float wr_s[HDIM * HDIM];
    __shared__ float b_s[HDIM];
    int t = threadIdx.x;
    for (int i = t; i < fin * HDIM; i += blockDim.x) wr_s[i] = Wr[i];
    if (t < HDIM) b_s[t] = b[t];
    __syncthreads();
    int n = blockIdx.x * blockDim.x + t;
    if (n >= Ncnt) return;
    float acc[HDIM];
    #pragma unroll
    for (int c = 0; c < HDIM; ++c) acc[c] = agg[(size_t)n * HDIM + c] + b_s[c];
    for (int f = 0; f < fin; ++f) {
        float hv = hin[(size_t)n * fin + f];
        #pragma unroll
        for (int c = 0; c < HDIM; ++c) acc[c] += hv * wr_s[f * HDIM + c];
    }
    #pragma unroll
    for (int c = 0; c < HDIM; ++c) hout[(size_t)n * HDIM + c] = leaky(acc[c]);
}

// ---------------- global sum pool ----------------
__global__ void pool_kernel(const float* __restrict__ h, const int* __restrict__ seg,
                            float* __restrict__ g, int Ncnt) {
    int i = blockIdx.x * blockDim.x + threadIdx.x;
    if (i >= Ncnt * HDIM) return;
    int n = i / HDIM, c = i % HDIM;
    atomicAdd(&g[(size_t)seg[n] * HDIM + c], h[i]);
}

// ---------------- dense head + VAE reparameterization ----------------
__global__ __launch_bounds__(256)
void head_kernel(const float* __restrict__ g, const float* __restrict__ eps,
                 const float* __restrict__ Wd1, const float* __restrict__ bd1,
                 const float* __restrict__ Wd2, const float* __restrict__ bd2,
                 const float* __restrict__ Wmu, const float* __restrict__ bmu,
                 const float* __restrict__ Wlv, const float* __restrict__ blv,
                 float* __restrict__ out) {
    __shared__ float gv[HDIM];
    __shared__ float d1[256];
    __shared__ float d2[256];
    const int gi = blockIdx.x, t = threadIdx.x;
    if (t < HDIM) gv[t] = g[(size_t)gi * HDIM + t];
    __syncthreads();
    float a = bd1[t];
    #pragma unroll
    for (int f = 0; f < HDIM; ++f) a += gv[f] * Wd1[f * 256 + t];
    d1[t] = leaky(a);
    __syncthreads();
    a = bd2[t];
    for (int f = 0; f < 256; ++f) a += d1[f] * Wd2[f * 256 + t];
    d2[t] = leaky(a);
    __syncthreads();
    if (t < LATD) {
        float mu = bmu[t], lv = blv[t];
        for (int f = 0; f < 256; ++f) {
            mu += d2[f] * Wmu[f * LATD + t];
            lv += d2[f] * Wlv[f * LATD + t];
        }
        out[(size_t)gi * LATD + t] = mu + __expf(0.5f * lv) * eps[(size_t)gi * LATD + t];
    }
}

// ---------------- host orchestration ----------------
extern "C" void kernel_launch(void* const* d_in, const int* in_sizes, int n_in,
                              void* d_out, int out_size, void* d_ws, size_t ws_size,
                              hipStream_t stream) {
    const float* x     = (const float*)d_in[0];
    const float* e     = (const float*)d_in[1];
    const int*   src   = (const int*)  d_in[2];
    const int*   tgt   = (const int*)  d_in[3];
    const int*   seg   = (const int*)  d_in[4];
    const float* eps   = (const float*)d_in[5];
    const float* W_pre = (const float*)d_in[6];
    const float* b_pre = (const float*)d_in[7];
    const float* Wk[3] = {(const float*)d_in[8],  (const float*)d_in[12], (const float*)d_in[16]};
    const float* bk[3] = {(const float*)d_in[9],  (const float*)d_in[13], (const float*)d_in[17]};
    const float* Wr[3] = {(const float*)d_in[10], (const float*)d_in[14], (const float*)d_in[18]};
    const float* bb[3] = {(const float*)d_in[11], (const float*)d_in[15], (const float*)d_in[19]};
    const float* W_d1 = (const float*)d_in[20]; const float* b_d1 = (const float*)d_in[21];
    const float* W_d2 = (const float*)d_in[22]; const float* b_d2 = (const float*)d_in[23];
    const float* W_mu = (const float*)d_in[24]; const float* b_mu = (const float*)d_in[25];
    const float* W_lv = (const float*)d_in[26]; const float* b_lv = (const float*)d_in[27];

    const int N = in_sizes[4];                 // seg length
    const int E = in_sizes[2];                 // src length
    const int G = in_sizes[5] / LATD;          // eps rows

    // workspace layout (256B aligned)
    unsigned char* ws = (unsigned char*)d_ws;
    size_t off = 0;
    auto alloc = [&](size_t bytes) {
        void* p = ws + off;
        off += (bytes + 255) & ~(size_t)255;
        return p;
    };
    float* h0  = (float*)alloc((size_t)N * PREW * 4);
    float* hA  = (float*)alloc((size_t)N * HDIM * 4);
    float* hB  = (float*)alloc((size_t)N * HDIM * 4);
    float* agg = (float*)alloc((size_t)N * HDIM * 4);
    float* g   = (float*)alloc((size_t)G * HDIM * 4);
    const int finLog2_l[3] = {4, 5, 5};        // fin = 16, 32, 32
    const int nChunks_l[3] = {4, 7, 7};        // Kpad = (S+1)*fin padded to x32: 128, 224, 224
    unsigned short* Wfrag[3];
    for (int l = 0; l < 3; ++l)
        Wfrag[l] = (unsigned short*)alloc((size_t)2 * nChunks_l[l] * 32 * 16 * 2);

    const int TB = 256;
    // build bf16 pre-permuted weights (one-time, tiny)
    for (int l = 0; l < 3; ++l) {
        int n = 2 * nChunks_l[l] * 32 * 16;
        build_wb<<<(n + TB - 1) / TB, TB, 0, stream>>>(Wk[l], bk[l], Wfrag[l],
                                                       1 << finLog2_l[l], nChunks_l[l]);
    }
    // precondition
    pre_kernel<<<(N + TB - 1) / TB, TB, 0, stream>>>(x, W_pre, b_pre, h0, N);

    // three ECC layers (ping-pong h0 -> hA -> hB -> hA)
    const float* layer_in[3]  = {h0, hA, hB};
    float*       layer_out[3] = {hA, hB, hA};
    const int edge_blocks = (E + 63) / 64;
    for (int l = 0; l < 3; ++l) {
        int nAgg = N * HDIM;
        int fin  = 1 << finLog2_l[l];
        int Kpad = nChunks_l[l] << 5;
        zero_f32<<<(nAgg + TB - 1) / TB, TB, 0, stream>>>(agg, nAgg);
        size_t smem = (size_t)64 * Kpad * 2 + (size_t)64 * fin * 4
                    + (size_t)64 * SDIM * 4 + 64 * 4;
        ecc_edge_wmma<<<edge_blocks, 128, smem, stream>>>(
            layer_in[l], finLog2_l[l], nChunks_l[l], e, src, tgt, Wfrag[l], agg, E);
        node_update<<<(N + TB - 1) / TB, TB, 0, stream>>>(
            layer_in[l], agg, Wr[l], bb[l], layer_out[l], fin, N);
    }
    // pool + head
    {
        int nG = G * HDIM;
        zero_f32<<<(nG + TB - 1) / TB, TB, 0, stream>>>(g, nG);
        int nP = N * HDIM;
        pool_kernel<<<(nP + TB - 1) / TB, TB, 0, stream>>>(hA, seg, g, N);
        head_kernel<<<G, 256, 0, stream>>>(g, eps, W_d1, b_d1, W_d2, b_d2,
                                           W_mu, b_mu, W_lv, b_lv, (float*)d_out);
    }
    (void)n_in; (void)out_size; (void)ws_size;
}
